// DecoderBlock_27779848471543
// MI455X (gfx1250) — matrix operand order repaired and checked by reading, not compiled, not run on hardware
//
#include <hip/hip_runtime.h>
#include <hip/hip_bf16.h>
#include <stdint.h>

typedef unsigned short u16;
typedef unsigned int   u32;
typedef __attribute__((ext_vector_type(8)))  __bf16 v8bf;
typedef __attribute__((ext_vector_type(16))) __bf16 v16bf;
typedef __attribute__((ext_vector_type(8)))  float  v8f;
typedef __attribute__((ext_vector_type(4)))  u32    u32x4;
typedef __attribute__((ext_vector_type(8)))  u32    u32x8;

#define B_   4
#define T_   2048
#define E_   1024
#define H_   16
#define D_   64
#define FF_  4096
#define BT_  8192   // B*T

// ---------- helpers ----------

__device__ __forceinline__ u16 f2bf(float f) {
    u32 u = __float_as_uint(f);
    u32 r = (u + 0x7FFFu + ((u >> 16) & 1u)) >> 16;   // RNE
    return (u16)r;
}

// Build a v16bf fragment from two 16-byte chunks (CDNA5 A/B frag halves).
__device__ __forceinline__ v16bf ld_frag2(const u16* p0, const u16* p1) {
    v8bf lo = *(const v8bf*)p0;
    v8bf hi = *(const v8bf*)p1;
    v16bf r;
#pragma unroll
    for (int i = 0; i < 8; ++i) { r[i] = lo[i]; r[i + 8] = hi[i]; }
    return r;
}
__device__ __forceinline__ v16bf ld_frag1(const u16* p) { return ld_frag2(p, p + 8); }

__device__ __forceinline__ v8f v8f_zero() {
    v8f z = {0.f, 0.f, 0.f, 0.f, 0.f, 0.f, 0.f, 0.f};
    return z;
}

// ---------- Tensor Data Mover: 2D bf16 tile global -> LDS ----------
// D# group0: count=1 | lds_addr | global_addr(57b) | type=2
// D# group1: data_size=2B, tensor_dim0/1, tile 32 x tile_d1, dim0 stride (elements)
__device__ __forceinline__ void tdm_load_2d(u32 lds_addr, const u16* gptr,
                                            u32 tile_d0, u32 tile_d1,
                                            u32 tensor_d0, u32 tensor_d1,
                                            u32 stride_elems) {
    unsigned long long ga = (unsigned long long)(size_t)gptr;
    u32x4 g0;
    g0[0] = 1u;                                        // count=1 (valid user D#)
    g0[1] = lds_addr;                                  // LDS byte address
    g0[2] = (u32)(ga & 0xFFFFFFFFu);                   // global_addr[31:0]
    g0[3] = (u32)((ga >> 32) & 0x1FFFFFFu) | (2u << 30); // global_addr[56:32] | type=2
    u32x8 g1;
    g1[0] = (1u << 16);                                // data_size=1 -> 2 bytes; mask=0
    g1[1] = (tensor_d0 & 0xFFFFu) << 16;               // tensor_dim0[15:0]
    g1[2] = (tensor_d0 >> 16) | ((tensor_d1 & 0xFFFFu) << 16);
    g1[3] = (tensor_d1 >> 16) | (tile_d0 << 16);       // tensor_dim1[31:16] | tile_dim0
    g1[4] = tile_d1 & 0xFFFFu;                         // tile_dim1 (tile_dim2=0)
    g1[5] = stride_elems;                              // tensor_dim0_stride[31:0]
    g1[6] = 0;                                         // stride[47:32] | dim1_stride[15:0]
    g1[7] = 0;
    asm volatile("tensor_load_to_lds %0, %1" :: "s"(g0), "s"(g1) : "memory");
}

// ---------- weight conversion ----------

// wq/wk/wv are [H][E][D] fp32.  Output: [3*H*D][E] bf16 (i.e. W^T, K-contiguous)
__global__ __launch_bounds__(256) void cvt_qkv_kernel(const float* __restrict__ wq,
                                                      const float* __restrict__ wk,
                                                      const float* __restrict__ wv,
                                                      u16* __restrict__ out) {
    int idx = blockIdx.x * 256 + threadIdx.x;        // over 3072*1024
    int n = idx >> 10;                               // 0..3071 output row
    int e = idx & 1023;                              // K index
    int m = n >> 10;                                 // which matrix
    int nn = n & 1023;
    int h = nn >> 6, d = nn & 63;
    const float* src = (m == 0) ? wq : (m == 1) ? wk : wv;
    out[idx] = f2bf(src[((size_t)h * E_ + e) * D_ + d]);
}

// in: [K][N] fp32 row-major   ->   out: [N][K] bf16
__global__ __launch_bounds__(256) void cvt_t_kernel(const float* __restrict__ in,
                                                    u16* __restrict__ out,
                                                    int K, int N) {
    size_t idx = (size_t)blockIdx.x * 256 + threadIdx.x;
    int n = (int)(idx / (size_t)K);
    int k = (int)(idx % (size_t)K);
    out[idx] = f2bf(in[(size_t)k * N + n]);
}

// ---------- layernorm (fp32 in -> bf16 out) ----------

__global__ __launch_bounds__(256) void ln_kernel(const float* __restrict__ x,
                                                 const float* __restrict__ g,
                                                 const float* __restrict__ beta,
                                                 u16* __restrict__ out) {
    __shared__ float s1[256], s2[256];
    int row = blockIdx.x, tid = threadIdx.x;
    const float* xr = x + (size_t)row * E_;
    float v[4], s = 0.f, q = 0.f;
#pragma unroll
    for (int i = 0; i < 4; ++i) {
        v[i] = xr[tid + i * 256];
        s += v[i]; q += v[i] * v[i];
    }
    s1[tid] = s; s2[tid] = q;
    __syncthreads();
    for (int off = 128; off > 0; off >>= 1) {
        if (tid < off) { s1[tid] += s1[tid + off]; s2[tid] += s2[tid + off]; }
        __syncthreads();
    }
    float mu  = s1[0] * (1.0f / E_);
    float var = s2[0] * (1.0f / E_) - mu * mu;
    float inv = rsqrtf(var + 1e-5f);
    u16* orow = out + (size_t)row * E_;
#pragma unroll
    for (int i = 0; i < 4; ++i) {
        int c = tid + i * 256;
        orow[c] = f2bf((v[i] - mu) * inv * g[c] + beta[c]);
    }
}

// ---------- generic WMMA bf16 GEMM with TDM tile feeding ----------
// C[M][N] = A[M][K](bf16) * Bt[N][K](bf16)  (+bias) (ReLU) (+resid), out fp32 or bf16.
// Block tile 128x128, 8 waves in 2(M) x 4(N) grid, wave tile 64x32, K-step 32.
// Wave0 DMAs the A tile, wave1 the B tile (tensor_load_to_lds, TENSORcnt).

template <bool BIAS, bool RESID, bool RELU, bool OUTBF>
__global__ __launch_bounds__(256) void gemm_kernel(const u16* __restrict__ A,
                                                   const u16* __restrict__ Bt,
                                                   const float* __restrict__ bias,
                                                   const float* __restrict__ resid,
                                                   float* __restrict__ Cf,
                                                   u16* __restrict__ Cb,
                                                   int M, int N, int K) {
    __shared__ __align__(16) u16 lA[2][128][32];
    __shared__ __align__(16) u16 lB[2][128][32];

    int tid = threadIdx.x, lane = tid & 31, w = tid >> 5;
    int wm = w >> 2, wn = w & 3;
    int bm = blockIdx.y * 128, bn = blockIdx.x * 128;

    v8f acc[4][2];
#pragma unroll
    for (int i = 0; i < 4; ++i)
#pragma unroll
        for (int j = 0; j < 2; ++j) acc[i][j] = v8f_zero();

    // issue TDM copies for k-tile kt into LDS buffer b (wave0: A, wave1: B)
    auto issue_tiles = [&](int kt, int b) {
        if (w == 0) {
            tdm_load_2d((u32)(size_t)&lA[b][0][0],
                        A + (size_t)bm * K + (size_t)kt * 32,
                        /*tile*/ 32, 128, /*tensor*/ (u32)K, 128u, (u32)K);
        } else if (w == 1) {
            tdm_load_2d((u32)(size_t)&lB[b][0][0],
                        Bt + (size_t)bn * K + (size_t)kt * 32,
                        /*tile*/ 32, 128, /*tensor*/ (u32)K, 128u, (u32)K);
        }
    };

    issue_tiles(0, 0);
    if (w < 2) __builtin_amdgcn_s_wait_tensorcnt(0);
    __syncthreads();

    int KT = K >> 5;
    int buf = 0;
    for (int kt = 0; kt < KT; ++kt) {
        if (kt + 1 < KT) issue_tiles(kt + 1, buf ^ 1);   // async DMA next tile

        int kh = lane >> 4, rr = lane & 15;
        v16bf afr[4];
#pragma unroll
        for (int mi = 0; mi < 4; ++mi) {
            const u16* pr = &lA[buf][wm * 64 + mi * 16 + rr][0];
            afr[mi] = ld_frag2(pr + kh * 8, pr + 16 + kh * 8);
        }
#pragma unroll
        for (int ni = 0; ni < 2; ++ni) {
            const u16* pb = &lB[buf][wn * 32 + ni * 16 + rr][0];
            v16bf bfr = ld_frag2(pb + kh * 8, pb + 16 + kh * 8);
#pragma unroll
            for (int mi = 0; mi < 4; ++mi) {
                acc[mi][ni] = __builtin_amdgcn_wmma_f32_16x16x32_bf16(
                    false, afr[mi], false, bfr, (short)0, acc[mi][ni], false, false);
            }
        }

        if (kt + 1 < KT) {
            if (w < 2) __builtin_amdgcn_s_wait_tensorcnt(0);
            __syncthreads();   // next buffer ready; everyone done with current
            buf ^= 1;
        }
    }

    // epilogue: C layout row = r + 8*(lane>>4), col = lane&15 within each 16x16 tile
    int eh = lane >> 4, ec = lane & 15;
#pragma unroll
    for (int mi = 0; mi < 4; ++mi) {
#pragma unroll
        for (int ni = 0; ni < 2; ++ni) {
#pragma unroll
            for (int r = 0; r < 8; ++r) {
                int row = bm + wm * 64 + mi * 16 + r + 8 * eh;
                int col = bn + wn * 32 + ni * 16 + ec;
                float v = acc[mi][ni][r];
                if (BIAS)  v += bias[col];
                if (RELU)  v = fmaxf(v, 0.0f);
                if (RESID) v += resid[(size_t)row * N + col];
                size_t o = (size_t)row * N + col;
                if (OUTBF) Cb[o] = f2bf(v);
                else       Cf[o] = v;
            }
        }
    }
}

// ---------- flash attention (causal) ----------
// qkv: [B*T][3*E] bf16 (q | k | v, each head-blocked h*64+d). attno: [B*T][E] bf16.
// One block = one (b,h) and 128 query rows; 8 waves x 16 rows each; key tiles of 64.

__global__ __launch_bounds__(256) void attn_kernel(const u16* __restrict__ qkv,
                                                   u16* __restrict__ attno) {
    __shared__ __align__(16) u16 Ksh[64][64];       // K rows [s][d]
    __shared__ __align__(16) u16 Vt[64][64];        // V^T    [d][s]
    __shared__ __align__(16) u16 Psh[8][16][64];    // per-wave P staging

    int tid = threadIdx.x, lane = tid & 31, w = tid >> 5;
    int kh = lane >> 4, l16 = lane & 15;
    int bh = blockIdx.y;
    int b = bh >> 4, hh = bh & 15;
    int q0 = blockIdx.x * 128;
    int qrow = q0 + w * 16 + l16;                   // A-frag row (t index)

    // Q fragments for this wave's 16 rows (D=64 -> 2 k-chunks of 32)
    const u16* qbase = qkv + ((size_t)(b * T_ + qrow) * (3 * E_)) + hh * D_;
    v16bf qf[2];
#pragma unroll
    for (int c = 0; c < 2; ++c)
        qf[c] = ld_frag2(qbase + c * 32 + kh * 8, qbase + c * 32 + 16 + kh * 8);

    v8f Oacc[4];
#pragma unroll
    for (int i = 0; i < 4; ++i) Oacc[i] = v8f_zero();
    float m_[8], l_[8];
#pragma unroll
    for (int r = 0; r < 8; ++r) { m_[r] = -1e30f; l_[r] = 0.f; }

    const float scale = 0.03125f;                   // E^-0.5 = 1/32 (per reference)
    int rowt_base = q0 + w * 16 + 8 * kh;           // + r = this lane's row t

    int nkb = (q0 >> 6) + 2;                        // causal: keys up to q0+127
    for (int kb = 0; kb < nkb; ++kb) {
        __syncthreads();
        // cooperative load of K tile and transposed V tile
        {
            int rr = tid >> 2, seg = (tid & 3) * 16;
            const u16* kp = qkv + ((size_t)(b * T_ + kb * 64 + rr) * (3 * E_)) + E_ + hh * D_ + seg;
            uint4 k0 = *(const uint4*)kp;
            uint4 k1 = *(const uint4*)(kp + 8);
            *(uint4*)&Ksh[rr][seg]     = k0;
            *(uint4*)&Ksh[rr][seg + 8] = k1;
            const u16* vp = qkv + ((size_t)(b * T_ + kb * 64 + rr) * (3 * E_)) + 2 * E_ + hh * D_ + seg;
#pragma unroll
            for (int i = 0; i < 16; ++i) Vt[seg + i][rr] = vp[i];
        }
        __syncthreads();

        // S = Q * K^T  (16 x 64 per wave)
        v8f S[4];
#pragma unroll
        for (int i = 0; i < 4; ++i) S[i] = v8f_zero();
#pragma unroll
        for (int c = 0; c < 2; ++c) {
#pragma unroll
            for (int nt = 0; nt < 4; ++nt) {
                const u16* pb = &Ksh[nt * 16 + l16][c * 32];
                v16bf bfr = ld_frag2(pb + kh * 8, pb + 16 + kh * 8);
                S[nt] = __builtin_amdgcn_wmma_f32_16x16x32_bf16(
                    false, qf[c], false, bfr, (short)0, S[nt], false, false);
            }
        }

        // scale + causal mask
#pragma unroll
        for (int nt = 0; nt < 4; ++nt) {
            int sglob = kb * 64 + nt * 16 + l16;
#pragma unroll
            for (int r = 0; r < 8; ++r) {
                float v = S[nt][r] * scale;
                if (sglob > rowt_base + r) v = -1e30f;
                S[nt][r] = v;
            }
        }

        // online softmax row stats (reduce across 16-lane half-wave)
        float mnew[8], alpha[8];
#pragma unroll
        for (int r = 0; r < 8; ++r) {
            float mx = fmaxf(fmaxf(S[0][r], S[1][r]), fmaxf(S[2][r], S[3][r]));
#pragma unroll
            for (int off = 1; off < 16; off <<= 1)
                mx = fmaxf(mx, __shfl_xor(mx, off, 32));
            mnew[r]  = fmaxf(m_[r], mx);
            alpha[r] = __expf(m_[r] - mnew[r]);
        }
        float rs[8];
#pragma unroll
        for (int r = 0; r < 8; ++r) rs[r] = 0.f;
#pragma unroll
        for (int nt = 0; nt < 4; ++nt) {
#pragma unroll
            for (int r = 0; r < 8; ++r) {
                float p = __expf(S[nt][r] - mnew[r]);
                S[nt][r] = p;
                rs[r] += p;
            }
        }
#pragma unroll
        for (int r = 0; r < 8; ++r) {
#pragma unroll
            for (int off = 1; off < 16; off <<= 1)
                rs[r] += __shfl_xor(rs[r], off, 32);
            l_[r] = l_[r] * alpha[r] + rs[r];
            m_[r] = mnew[r];
        }
#pragma unroll
        for (int dt = 0; dt < 4; ++dt)
#pragma unroll
            for (int r = 0; r < 8; ++r) Oacc[dt][r] *= alpha[r];

        // stage P (bf16) to per-wave LDS, then O += P * V via WMMA
#pragma unroll
        for (int nt = 0; nt < 4; ++nt)
#pragma unroll
            for (int r = 0; r < 8; ++r)
                Psh[w][r + 8 * kh][nt * 16 + l16] = f2bf(S[nt][r]);

#pragma unroll
        for (int c = 0; c < 2; ++c) {
            const u16* pr = &Psh[w][l16][0];
            v16bf af = ld_frag2(pr + c * 32 + kh * 8, pr + c * 32 + 16 + kh * 8);
#pragma unroll
            for (int dt = 0; dt < 4; ++dt) {
                const u16* pb = &Vt[dt * 16 + l16][c * 32];
                v16bf bfr = ld_frag2(pb + kh * 8, pb + 16 + kh * 8);
                Oacc[dt] = __builtin_amdgcn_wmma_f32_16x16x32_bf16(
                    false, af, false, bfr, (short)0, Oacc[dt], false, false);
            }
        }
    }

    // normalize + write concat-head output (bf16)
#pragma unroll
    for (int dt = 0; dt < 4; ++dt) {
#pragma unroll
        for (int r = 0; r < 8; ++r) {
            int t = q0 + w * 16 + r + 8 * kh;
            int col = hh * D_ + dt * 16 + l16;
            float v = Oacc[dt][r] / l_[r];
            attno[(size_t)(b * T_ + t) * E_ + col] = f2bf(v);
        }
    }
}

// ---------- launch ----------

extern "C" void kernel_launch(void* const* d_in, const int* in_sizes, int n_in,
                              void* d_out, int out_size, void* d_ws, size_t ws_size,
                              hipStream_t stream) {
    const float* x     = (const float*)d_in[0];
    const float* wq    = (const float*)d_in[1];
    const float* wk    = (const float*)d_in[2];
    const float* wv    = (const float*)d_in[3];
    const float* projw = (const float*)d_in[4];
    const float* projb = (const float*)d_in[5];
    const float* g1    = (const float*)d_in[6];
    const float* be1   = (const float*)d_in[7];
    const float* g2    = (const float*)d_in[8];
    const float* be2   = (const float*)d_in[9];
    const float* w1    = (const float*)d_in[10];
    const float* bf1   = (const float*)d_in[11];
    const float* w2    = (const float*)d_in[12];
    const float* bf2   = (const float*)d_in[13];
    float* out = (float*)d_out;

    char* p = (char*)d_ws;
    u16*   h1    = (u16*)p;   p += (size_t)BT_ * E_ * 2;        // ln1(x) bf16
    u16*   qkv   = (u16*)p;   p += (size_t)BT_ * 3 * E_ * 2;    // q|k|v bf16
    u16*   attno = (u16*)p;   p += (size_t)BT_ * E_ * 2;        // attention out bf16
    float* x1    = (float*)p; p += (size_t)BT_ * E_ * 4;        // x + proj (fp32)
    u16*   h2    = (u16*)p;   p += (size_t)BT_ * E_ * 2;        // ln2(x1) bf16
    u16*   ffn1  = (u16*)p;   p += (size_t)BT_ * FF_ * 2;       // relu(h2 w1 + b) bf16
    u16*   wqkvT = (u16*)p;   p += (size_t)3 * E_ * E_ * 2;
    u16*   projT = (u16*)p;   p += (size_t)E_ * E_ * 2;
    u16*   w1T   = (u16*)p;   p += (size_t)E_ * FF_ * 2;
    u16*   w2T   = (u16*)p;   p += (size_t)FF_ * E_ * 2;

    // weight conversion (fp32 -> bf16, transposed to [N][K])
    cvt_qkv_kernel<<<(3 * E_ * E_) / 256, 256, 0, stream>>>(wq, wk, wv, wqkvT);
    cvt_t_kernel  <<<(E_ * E_) / 256,  256, 0, stream>>>(projw, projT, E_, E_);
    cvt_t_kernel  <<<(E_ * FF_) / 256, 256, 0, stream>>>(w1, w1T, E_, FF_);
    cvt_t_kernel  <<<(FF_ * E_) / 256, 256, 0, stream>>>(w2, w2T, FF_, E_);

    // ln1 -> QKV -> attention
    ln_kernel<<<BT_, 256, 0, stream>>>(x, g1, be1, h1);
    gemm_kernel<false, false, false, true><<<dim3(3 * E_ / 128, BT_ / 128), 256, 0, stream>>>(
        h1, wqkvT, nullptr, nullptr, nullptr, qkv, BT_, 3 * E_, E_);
    attn_kernel<<<dim3(T_ / 128, B_ * H_), 256, 0, stream>>>(qkv, attno);

    // projection + residual (fp32 out)
    gemm_kernel<true, true, false, false><<<dim3(E_ / 128, BT_ / 128), 256, 0, stream>>>(
        attno, projT, projb, x, x1, nullptr, BT_, E_, E_);

    // ln2 -> FFN
    ln_kernel<<<BT_, 256, 0, stream>>>(x1, g2, be2, h2);
    gemm_kernel<true, false, true, true><<<dim3(FF_ / 128, BT_ / 128), 256, 0, stream>>>(
        h2, w1T, bf1, nullptr, nullptr, ffn1, BT_, FF_, E_);
    gemm_kernel<true, true, false, false><<<dim3(E_ / 128, BT_ / 128), 256, 0, stream>>>(
        ffn1, w2T, bf2, x1, out, nullptr, BT_, E_, FF_);
}
